// MultiHeadAttention_9363028705582
// MI455X (gfx1250) — matrix operand order repaired and checked by reading, not compiled, not run on hardware
//
#include <hip/hip_runtime.h>

typedef __attribute__((ext_vector_type(16))) _Float16 v16h;
typedef __attribute__((ext_vector_type(8)))  float    v8f;

#define B_  4
#define L_  2048
#define D_  1024
#define H_  16

struct __align__(16) h8 { _Float16 h[8]; };
struct __align__(8)  h4 { _Float16 h[4]; };

// ---- gfx1250 async global->LDS copy (ASYNCcnt-tracked), 16B per lane ----
__device__ inline void async_copy_b128(const void* gsrc, void* ldst) {
    unsigned lds = (unsigned)(uintptr_t)ldst;   // low 32 bits of flat LDS addr = LDS offset
    asm volatile("global_load_async_to_lds_b128 %0, %1, off"
                 :: "v"(lds), "v"(gsrc) : "memory");
}
__device__ inline void wait_async0() { asm volatile("s_wait_asynccnt 0x0" ::: "memory"); }
__device__ inline void wait_async8() { asm volatile("s_wait_asynccnt 0x8" ::: "memory"); }

// ---- WMMA operand gathers for V_WMMA_F32_16X16X32_F16 (wave32) ----
__device__ inline v16h load_wmma_a(const _Float16* src, int ldr, int kcol, int lane) {
    int m = lane & 15, half = lane >> 4;
    v16h a;
#pragma unroll
    for (int v = 0; v < 8; ++v) {
        int k = (v < 4) ? (half * 8 + 2 * v) : (16 + half * 8 + 2 * (v - 4));
        const _Float16* p = src + m * ldr + kcol + k;
        a[2 * v]     = p[0];
        a[2 * v + 1] = p[1];
    }
    return a;
}
// B is 32(K) x 16(N); source stored N-major: element (c,n) at src[n*ldr + ccol + c].
__device__ inline v16h load_wmma_b(const _Float16* src, int ldr, int ccol, int lane) {
    int n = lane & 15, half = lane >> 4;
    v16h b;
#pragma unroll
    for (int v = 0; v < 8; ++v) {
        int c = (v < 4) ? (half * 8 + 2 * v) : (16 + half * 8 + 2 * (v - 4));
        const _Float16* p = src + n * ldr + ccol + c;
        b[2 * v]     = p[0];
        b[2 * v + 1] = p[1];
    }
    return b;
}

// ---- Tiled GEMM: out(M x 1024) = act(M x 1024) * W(1024 x 1024) + bias ----
// mode 0: f16 result scattered to [B][H][L][64] (projections, act = fp32)
// mode 2: fp32 result + residual (output proj, act = f16 via async staging)
__global__ __launch_bounds__(128)
void gemm_kernel(const float* __restrict__ actF, const _Float16* __restrict__ actH,
                 const float* __restrict__ W, const float* __restrict__ bias,
                 const float* __restrict__ residual,
                 _Float16* __restrict__ outH, float* __restrict__ outF, int mode) {
    __shared__ _Float16 As[64 * 64];   // [m][k]
    __shared__ _Float16 Wt[64 * 64];   // transposed: [n][k]
    const int n0 = blockIdx.x * 64;
    const int m0 = blockIdx.y * 64;
    const int t = threadIdx.x;
    const int wave = t >> 5, lane = t & 31;

    v8f acc[4] = {};
    for (int k0 = 0; k0 < 1024; k0 += 64) {
        if (actF) {
            float4 ab[8];                             // batch all loads, then one wait
#pragma unroll
            for (int e = 0; e < 8; ++e) {             // 1024 float4 chunks
                int ci = t + 128 * e;
                int r = ci >> 4, c4 = (ci & 15) << 2;
                ab[e] = *(const float4*)&actF[(size_t)(m0 + r) * 1024 + k0 + c4];
            }
#pragma unroll
            for (int e = 0; e < 8; ++e) {
                int ci = t + 128 * e;
                int r = ci >> 4, c4 = (ci & 15) << 2;
                h4 hv;
                hv.h[0] = (_Float16)ab[e].x; hv.h[1] = (_Float16)ab[e].y;
                hv.h[2] = (_Float16)ab[e].z; hv.h[3] = (_Float16)ab[e].w;
                *(h4*)&As[r * 64 + c4] = hv;
            }
        } else {                                      // f16 act: async copy, no cvt
#pragma unroll
            for (int e = 0; e < 4; ++e) {             // 512 h8 chunks
                int ci = t + 128 * e;
                int r = ci >> 3, c8 = (ci & 7) << 3;
                async_copy_b128(&actH[(size_t)(m0 + r) * 1024 + k0 + c8], &As[r * 64 + c8]);
            }
        }
        {
            float4 wb[8];                             // batch W loads too
#pragma unroll
            for (int e = 0; e < 8; ++e) {
                int ci = t + 128 * e;
                int r = ci >> 4, c4 = (ci & 15) << 2;
                wb[e] = *(const float4*)&W[(size_t)(k0 + r) * 1024 + n0 + c4];
            }
#pragma unroll
            for (int e = 0; e < 8; ++e) {             // transposed store
                int ci = t + 128 * e;
                int r = ci >> 4, c4 = (ci & 15) << 2;
                Wt[(c4 + 0) * 64 + r] = (_Float16)wb[e].x;
                Wt[(c4 + 1) * 64 + r] = (_Float16)wb[e].y;
                Wt[(c4 + 2) * 64 + r] = (_Float16)wb[e].z;
                Wt[(c4 + 3) * 64 + r] = (_Float16)wb[e].w;
            }
        }
        if (!actF) wait_async0();
        __syncthreads();
#pragma unroll
        for (int kc = 0; kc < 64; kc += 32) {         // gather all, then 4 WMMAs
            v16h a  = load_wmma_a(As + wave * 1024, 64, kc, lane);
            v16h b0 = load_wmma_b(Wt +    0, 64, kc, lane);
            v16h b1 = load_wmma_b(Wt + 1024, 64, kc, lane);
            v16h b2 = load_wmma_b(Wt + 2048, 64, kc, lane);
            v16h b3 = load_wmma_b(Wt + 3072, 64, kc, lane);
            acc[0] = __builtin_amdgcn_wmma_f32_16x16x32_f16(false, a, false, b0, (short)0, acc[0], false, false);
            acc[1] = __builtin_amdgcn_wmma_f32_16x16x32_f16(false, a, false, b1, (short)0, acc[1], false, false);
            acc[2] = __builtin_amdgcn_wmma_f32_16x16x32_f16(false, a, false, b2, (short)0, acc[2], false, false);
            acc[3] = __builtin_amdgcn_wmma_f32_16x16x32_f16(false, a, false, b3, (short)0, acc[3], false, false);
        }
        __syncthreads();
    }
    int nlane = lane & 15, half = lane >> 4;
#pragma unroll
    for (int nt = 0; nt < 4; ++nt) {
#pragma unroll
        for (int r = 0; r < 8; ++r) {
            int m = m0 + wave * 16 + r + 8 * half;    // C/D layout: VGPR r -> M=r / r+8
            int n = n0 + nt * 16 + nlane;
            float vv = acc[nt][r] + bias[n];
            if (mode == 0) {
                int bb = m >> 11, l = m & (L_ - 1);
                int h = n >> 6, dk = n & 63;
                outH[((((size_t)bb * H_ + h) * L_ + l) << 6) + dk] = (_Float16)vv;
            } else {
                outF[(size_t)m * 1024 + n] = vv + residual[(size_t)m * 1024 + n];
            }
        }
    }
}

// ---- Attention: one WG (4 waves) per (b, h, 16-query block) ----
// LDS: Sbuf f16 16x2048 (64KB) | Qs 2KB | Kbuf0/Kbuf1 16KB each (double-buffered
// async K staging; reused as Vt and fp32 O-reduction in phase 4) | fp32 scratch.
__global__ __launch_bounds__(128)
void attn_kernel(const _Float16* __restrict__ qh, const _Float16* __restrict__ kh,
                 const _Float16* __restrict__ vh, const int* __restrict__ mask,
                 float* __restrict__ attn_out, _Float16* __restrict__ ctx) {
    extern __shared__ char smem[];
    _Float16* Sbuf  = (_Float16*)smem;                           // 65536 B
    _Float16* Qs    = (_Float16*)(smem + 65536);                 // 2048 B
    _Float16* Kbuf0 = (_Float16*)(smem + 65536 + 2048);          // 16384 B
    _Float16* Kbuf1 = (_Float16*)(smem + 65536 + 2048 + 16384);  // 16384 B
    float*    red   = (float*)(smem + 65536 + 2048 + 32768);     // 128 f32
    float*    Mrow  = red + 128;
    float*    Zrow  = Mrow + 16;

    const int qb = blockIdx.x, bh = blockIdx.y;
    const int b = bh >> 4, h = bh & 15;
    const int q0 = qb * 16;
    const int t = threadIdx.x, wave = t >> 5, lane = t & 31;
    const int nlane = lane & 15, half = lane >> 4;
    const size_t headBase = (size_t)bh * L_ * 64;

    // Q: single async 16B per thread (16x64 f16 contiguous)
    async_copy_b128(qh + headBase + ((size_t)q0 << 6) + t * 8, Qs + t * 8);
    wait_async0();
    __syncthreads();
    // Q is invariant for all of phase 1: hoist both A operands
    v16h a0  = load_wmma_a(Qs, 64, 0,  lane);
    v16h a32 = load_wmma_a(Qs, 64, 32, lane);

    auto issueK = [&](int kb, _Float16* buf) {
        const _Float16* src = kh + headBase + ((size_t)kb << 13);  // kb*128*64
#pragma unroll
        for (int e = 0; e < 8; ++e) {
            int off = (t + 128 * e) << 3;    // f16 units, 16B chunks
            async_copy_b128(src + off, buf + off);
        }
    };

    // Phase 1: S = Q K^T * 0.125, mask -> Sbuf (f16); double-buffered async K
    const float scale = 0.125f;
    _Float16* Kb[2] = {Kbuf0, Kbuf1};
    issueK(0, Kbuf0);
    for (int kb = 0; kb < 16; ++kb) {
        if (kb < 15) { issueK(kb + 1, Kb[(kb + 1) & 1]); wait_async8(); }
        else         { wait_async0(); }
        __syncthreads();
        const _Float16* Ks = Kb[kb & 1];
        const int kbase = kb * 128;
#pragma unroll
        for (int tt = 0; tt < 2; ++tt) {
            int tileoff = (wave * 2 + tt) * 16;
            int mv[8];                                 // batch mask loads up front
#pragma unroll
            for (int r = 0; r < 8; ++r) {
                int m = r + 8 * half;
                mv[r] = mask[(size_t)b * L_ * L_ + (size_t)(q0 + m) * L_ +
                             kbase + tileoff + nlane];
            }
            v16h b0 = load_wmma_b(Ks + tileoff * 64, 64, 0,  lane);
            v16h b1 = load_wmma_b(Ks + tileoff * 64, 64, 32, lane);
            v8f acc = {};
            acc = __builtin_amdgcn_wmma_f32_16x16x32_f16(false, a0,  false, b0, (short)0, acc, false, false);
            acc = __builtin_amdgcn_wmma_f32_16x16x32_f16(false, a32, false, b1, (short)0, acc, false, false);
#pragma unroll
            for (int r = 0; r < 8; ++r) {
                int m = r + 8 * half;
                int key = kbase + tileoff + nlane;
                float s = acc[r] * scale;
                if (mv[r] == 0) s = -30000.0f;  // exp underflows to 0 like ref's -1e10
                Sbuf[m * L_ + key] = (_Float16)s;
            }
        }
        __syncthreads();
    }

    // Phase 2+3: softmax per 2048-wide row; 8 threads/row, contiguous 256-col chunks,
    // all LDS traffic as 16B h8 vectors; normalize in place (same-address f16 write).
    {
        int r = t >> 3, g = t & 7;
        h8* srow = (h8*)(Sbuf + r * L_);          // 256 chunks per row
        const int i0 = g * 32, i1 = i0 + 32;
        float mx = -3.4e38f;
        for (int i = i0; i < i1; ++i) {
            h8 vch = srow[i];
#pragma unroll
            for (int j = 0; j < 8; ++j) mx = fmaxf(mx, (float)vch.h[j]);
        }
        red[r * 8 + g] = mx;
        __syncthreads();
        if (g == 0) {
            float m2 = red[r * 8];
#pragma unroll
            for (int i = 1; i < 8; ++i) m2 = fmaxf(m2, red[r * 8 + i]);
            Mrow[r] = m2;
        }
        __syncthreads();
        float mr = Mrow[r];
        float sum = 0.f;
        for (int i = i0; i < i1; ++i) {
            h8 vch = srow[i];
#pragma unroll
            for (int j = 0; j < 8; ++j) sum += expf((float)vch.h[j] - mr);
        }
        red[r * 8 + g] = sum;
        __syncthreads();
        if (g == 0) {
            float z = 0.f;
#pragma unroll
            for (int i = 0; i < 8; ++i) z += red[r * 8 + i];
            Zrow[r] = 1.0f / z;
        }
        __syncthreads();
        float invz = Zrow[r];
        float4* arow4 = (float4*)(attn_out + (size_t)bh * L_ * L_ + (size_t)(q0 + r) * L_);
        for (int i = i0; i < i1; ++i) {
            h8 vch = srow[i];
            float p[8];
#pragma unroll
            for (int j = 0; j < 8; ++j) p[j] = expf((float)vch.h[j] - mr) * invz;
            h8 pch;
#pragma unroll
            for (int j = 0; j < 8; ++j) pch.h[j] = (_Float16)p[j];
            srow[i] = pch;
            float4 lo, hi;
            lo.x = p[0]; lo.y = p[1]; lo.z = p[2]; lo.w = p[3];
            hi.x = p[4]; hi.y = p[5]; hi.z = p[6]; hi.w = p[7];
            arow4[2 * i] = lo; arow4[2 * i + 1] = hi;
        }
    }
    __syncthreads();

    // Phase 4: O = P * V ; waves split keys (32 each per 128-key block)
    _Float16* Vt   = Kbuf0;          // [dv][key] 64x128
    float*    Ored = (float*)Kbuf1;  // 4 waves * 16*64 fp32 = 16KB
    v8f oacc[4] = {};
    for (int kb = 0; kb < 16; ++kb) {
        const int kbase = kb * 128;
        h8 vbuf[8];                                   // batch loads, then transpose
#pragma unroll
        for (int e = 0; e < 8; ++e) {                 // 1024 h8 chunks
            int ci = t + 128 * e;
            int key = ci >> 3, dv0 = (ci & 7) << 3;
            vbuf[e] = *(const h8*)&vh[headBase + ((size_t)(kbase + key) << 6) + dv0];
        }
#pragma unroll
        for (int e = 0; e < 8; ++e) {
            int ci = t + 128 * e;
            int key = ci >> 3, dv0 = (ci & 7) << 3;
#pragma unroll
            for (int j = 0; j < 8; ++j) Vt[(dv0 + j) * 128 + key] = vbuf[e].h[j];
        }
        __syncthreads();
        int chunk = wave * 32;
        v16h a  = load_wmma_a(Sbuf, L_, kbase + chunk, lane);
        v16h b0 = load_wmma_b(Vt +       0, 128, chunk, lane);
        v16h b1 = load_wmma_b(Vt + 16 * 128, 128, chunk, lane);
        v16h b2 = load_wmma_b(Vt + 32 * 128, 128, chunk, lane);
        v16h b3 = load_wmma_b(Vt + 48 * 128, 128, chunk, lane);
        oacc[0] = __builtin_amdgcn_wmma_f32_16x16x32_f16(false, a, false, b0, (short)0, oacc[0], false, false);
        oacc[1] = __builtin_amdgcn_wmma_f32_16x16x32_f16(false, a, false, b1, (short)0, oacc[1], false, false);
        oacc[2] = __builtin_amdgcn_wmma_f32_16x16x32_f16(false, a, false, b2, (short)0, oacc[2], false, false);
        oacc[3] = __builtin_amdgcn_wmma_f32_16x16x32_f16(false, a, false, b3, (short)0, oacc[3], false, false);
        __syncthreads();
    }

    float* myO = Ored + wave * 1024;
#pragma unroll
    for (int nt = 0; nt < 4; ++nt)
#pragma unroll
        for (int r = 0; r < 8; ++r)
            myO[(r + 8 * half) * 64 + nt * 16 + nlane] = oacc[nt][r];
    __syncthreads();
#pragma unroll
    for (int e = 0; e < 8; ++e) {
        int idx = t + 128 * e;                  // 16*64
        int r = idx >> 6, dv = idx & 63;
        float o = Ored[idx] + Ored[1024 + idx] + Ored[2048 + idx] + Ored[3072 + idx];
        ctx[((size_t)(b * L_ + q0 + r) << 10) + (h << 6) + dv] = (_Float16)o;
    }
}

// ---- LayerNorm over D=1024 per row ----
__global__ __launch_bounds__(256)
void layernorm_kernel(const float* __restrict__ x, const float* __restrict__ g,
                      const float* __restrict__ bb, float* __restrict__ out) {
    __shared__ float rs[256], rq[256];
    int row = blockIdx.x, t = threadIdx.x;
    const float* xr = x + (size_t)row * 1024;
    float s = 0.f, q = 0.f;
#pragma unroll
    for (int c = t; c < 1024; c += 256) { float v = xr[c]; s += v; q += v * v; }
    rs[t] = s; rq[t] = q;
    __syncthreads();
    for (int off = 128; off > 0; off >>= 1) {
        if (t < off) { rs[t] += rs[t + off]; rq[t] += rq[t + off]; }
        __syncthreads();
    }
    float mu  = rs[0] * (1.0f / 1024.0f);
    float var = rq[0] * (1.0f / 1024.0f) - mu * mu;
    float inv = rsqrtf(var + 1e-6f);
#pragma unroll
    for (int c = t; c < 1024; c += 256)
        out[(size_t)row * 1024 + c] = (xr[c] - mu) * inv * g[c] + bb[c];
}

extern "C" void kernel_launch(void* const* d_in, const int* in_sizes, int n_in,
                              void* d_out, int out_size, void* d_ws, size_t ws_size,
                              hipStream_t stream) {
    (void)in_sizes; (void)n_in; (void)out_size; (void)ws_size;
    const float* q    = (const float*)d_in[0];
    const float* k    = (const float*)d_in[1];
    const float* v    = (const float*)d_in[2];
    const int*   mask = (const int*)  d_in[3];
    const float* Wq   = (const float*)d_in[4];
    const float* bq   = (const float*)d_in[5];
    const float* Wk   = (const float*)d_in[6];
    const float* bk   = (const float*)d_in[7];
    const float* Wv   = (const float*)d_in[8];
    const float* bv   = (const float*)d_in[9];
    const float* Wo   = (const float*)d_in[10];
    const float* bo   = (const float*)d_in[11];
    const float* lng  = (const float*)d_in[12];
    const float* lnb  = (const float*)d_in[13];

    char* ws = (char*)d_ws;                                 // 96 MB used
    _Float16* qh  = (_Float16*)(ws);                        // 16 MB  [B][H][L][64]
    _Float16* kh  = (_Float16*)(ws + (size_t)16777216);     // 16 MB
    _Float16* vh  = (_Float16*)(ws + (size_t)2 * 16777216); // 16 MB
    _Float16* ctx = (_Float16*)(ws + (size_t)3 * 16777216); // 16 MB  [B*L][1024]
    float*    tmp = (float*)   (ws + (size_t)4 * 16777216); // 32 MB  [B*L][1024]

    float* out  = (float*)d_out;
    float* attn = out + (size_t)B_ * L_ * D_;               // second tuple output

    dim3 gblk(128);
    dim3 ggrid(16, 128);                                    // N/64, M/64
    gemm_kernel<<<ggrid, gblk, 0, stream>>>(q, nullptr, Wq, bq, nullptr, qh, nullptr, 0);
    gemm_kernel<<<ggrid, gblk, 0, stream>>>(k, nullptr, Wk, bk, nullptr, kh, nullptr, 0);
    gemm_kernel<<<ggrid, gblk, 0, stream>>>(v, nullptr, Wv, bv, nullptr, vh, nullptr, 0);

    size_t shmem = 65536 + 2048 + 2 * 16384 + (128 + 16 + 16) * sizeof(float);
    attn_kernel<<<dim3(L_ / 16, B_ * H_), dim3(128), shmem, stream>>>(
        qh, kh, vh, mask, attn, ctx);

    gemm_kernel<<<ggrid, gblk, 0, stream>>>(nullptr, ctx, Wo, bo, q, nullptr, tmp, 2);
    layernorm_kernel<<<dim3(B_ * L_), dim3(256), 0, stream>>>(tmp, lng, lnb, out);
}
